// Model_38835094290733
// MI455X (gfx1250) — matrix operand order repaired and checked by reading, not compiled, not run on hardware
//
#include <hip/hip_runtime.h>
#include <math.h>

// ---------------------------------------------------------------------------
// ConvMod block (NP=6 deformable branches + LN + MLP) for MI455X / gfx1250.
// All GEMMs run on v_wmma_f32_16x16x32_f16 (f16 operands, f32 accumulate).
// Register-blocked: each wave computes 16 tokens x 64 outputs (4 WMMA tiles),
// one A fragment + 4 distinct B fragments loaded up-front per k-chunk so the
// scheduler can clause the loads and use partial s_wait_loadcnt before WMMAs.
// B=8, C=64, L=1024 -> 8192 tokens. Everything is L2-resident (<=36MB bufs).
// ---------------------------------------------------------------------------

#define TOK   8192          // B*L tokens
#define DIM   64
#define LSEQ  1024
#define NPB   6
#define MLPH  256

typedef __attribute__((ext_vector_type(16))) _Float16 v16h;
typedef __attribute__((ext_vector_type(8)))  float    v8f;

// ------------------------- WMMA fragment loaders ---------------------------
// A matrix 16x32 f16 (tokens x k-chunk), row-major source [row, Kd]:
//   lane 0-15: M=lane, VGPR v holds K = (v>=4?16:0) + 2*(v&3) + h
//   lane16-31: M=lane-16, same + 8
__device__ __forceinline__ v16h load_A_frag(const _Float16* __restrict__ A,
                                            int lane, int t0, int Kd, int kc) {
    const _Float16* p = A + (size_t)(t0 + (lane & 15)) * Kd + kc + ((lane >> 4) & 1) * 8;
    union { v16h h; unsigned u[8]; } f;
#pragma unroll
    for (int v = 0; v < 8; ++v) {
        int k = ((v & 4) << 2) + ((v & 3) << 1);     // (v>=4?16:0) + 2*(v&3)
        f.u[v] = *(const unsigned*)(p + k);          // two consecutive halves
    }
    return f.h;
}

// B matrix 32x16 f16 (k-chunk x out), weight stored row-major [out, Kd]:
//   lane 0-15: N=lane, VGPR v holds K = (v>=4?16:0) + 2*(v&3) + h ; lane16-31: same + 8
__device__ __forceinline__ v16h load_B_frag(const _Float16* __restrict__ W,
                                            int lane, int n0, int Kd, int kc) {
    const _Float16* p = W + (size_t)(n0 + (lane & 15)) * Kd + kc + ((lane >> 4) & 1) * 8;
    union { v16h h; unsigned u[8]; } f;
#pragma unroll
    for (int v = 0; v < 8; ++v) f.u[v] = *(const unsigned*)(p + ((v & 4) << 2) + ((v & 3) << 1));
    return f.h;
}

__device__ __forceinline__ float gelu_exact(float x) {
    return 0.5f * x * (1.0f + erff(x * 0.70710678118654752f));
}

// ------------------------------- GEMM kernel -------------------------------
// out[row, col] = epi( sum_k A[row,k] * W[col,k] + bias[col] )
// block = 128 threads = 4 waves; wave w owns rows [blockIdx.x*64 + w*16, +16)
// and all 64 columns of n-band blockIdx.y (4 accumulators, shared A frag).
enum { EPI_F32 = 0, EPI_GELU_F16 = 1, EPI_ACC = 2, EPI_FINAL = 3 };

template <int EPI>
__global__ __launch_bounds__(128)
void gemm_wmma(const _Float16* __restrict__ A, const _Float16* __restrict__ W,
               const float* __restrict__ bias, int N, int Kd,
               float* __restrict__ outF, _Float16* __restrict__ outH,
               const float* __restrict__ x1, const float* __restrict__ gamma2)
{
    const int lane = threadIdx.x & 31;
    const int wave = threadIdx.x >> 5;
    const int t0 = blockIdx.x * 64 + wave * 16;
    const int n0 = blockIdx.y * 64;

    v8f acc0 = {}, acc1 = {}, acc2 = {}, acc3 = {};
    for (int kc = 0; kc < Kd; kc += 32) {
        // issue all fragment loads first (10x b128) so WMMAs can retire
        // against partial loadcnt waits instead of full drains
        const v16h a  = load_A_frag(A, lane, t0, Kd, kc);
        const v16h b0 = load_B_frag(W, lane, n0,      Kd, kc);
        const v16h b1 = load_B_frag(W, lane, n0 + 16, Kd, kc);
        const v16h b2 = load_B_frag(W, lane, n0 + 32, Kd, kc);
        const v16h b3 = load_B_frag(W, lane, n0 + 48, Kd, kc);
        acc0 = __builtin_amdgcn_wmma_f32_16x16x32_f16(false, a, false, b0,
                                                      (short)0, acc0, false, false);
        acc1 = __builtin_amdgcn_wmma_f32_16x16x32_f16(false, a, false, b1,
                                                      (short)0, acc1, false, false);
        acc2 = __builtin_amdgcn_wmma_f32_16x16x32_f16(false, a, false, b2,
                                                      (short)0, acc2, false, false);
        acc3 = __builtin_amdgcn_wmma_f32_16x16x32_f16(false, a, false, b3,
                                                      (short)0, acc3, false, false);
    }

    const int rbase = t0 + ((lane >> 4) & 1) * 8;
    v8f accs[4] = { acc0, acc1, acc2, acc3 };
#pragma unroll
    for (int j = 0; j < 4; ++j) {
        const int col = n0 + 16 * j + (lane & 15);
        const float bv = bias[col];
#pragma unroll
        for (int r = 0; r < 8; ++r) {
            const int row = rbase + r;
            float v = accs[j][r] + bv;
            if constexpr (EPI == EPI_F32) {
                outF[(size_t)row * N + col] = v;
            } else if constexpr (EPI == EPI_GELU_F16) {
                outH[(size_t)row * N + col] = (_Float16)gelu_exact(v);
            } else if constexpr (EPI == EPI_ACC) {
                outF[(size_t)row * N + col] += v;
            } else { // EPI_FINAL: N == 64; out = x1 + gamma2*mlp, scatter (B,C,L)
                float o = x1[row * DIM + col] + gamma2[col] * v;
                int b = row >> 10, l = row & 1023;
                outF[((size_t)b * DIM + col) * LSEQ + l] = o;
            }
        }
    }
}

// ---------------------------- elementwise kernels --------------------------
__global__ void prep_x(const float* __restrict__ x, float* __restrict__ xt,
                       _Float16* __restrict__ xth) {
    int i = blockIdx.x * blockDim.x + threadIdx.x;
    if (i >= TOK * DIM) return;
    int t = i >> 6, c = i & 63;
    int b = t >> 10, l = t & 1023;
    float v = x[((size_t)b * DIM + c) * LSEQ + l];
    xt[i] = v;
    xth[i] = (_Float16)v;
}

__global__ void cvt_f16(const float* __restrict__ s, _Float16* __restrict__ d, int n) {
    int i = blockIdx.x * blockDim.x + threadIdx.x;
    if (i < n) d[i] = (_Float16)s[i];
}

// src is (Kd, N) row-major; dst is (N, Kd) row-major f16
__global__ void tr_f16(const float* __restrict__ s, _Float16* __restrict__ d,
                       int Kd, int N) {
    int i = blockIdx.x * blockDim.x + threadIdx.x;
    if (i >= N * Kd) return;
    int n = i / Kd, k = i - n * Kd;
    d[i] = (_Float16)s[(size_t)k * N + n];
}

// DCN sampling: per (token, channel) softmax over K mask logits + bilinear
// interpolation of value at (l + k - center + offset), zero-padded.
__global__ void dcn_sample(const float* __restrict__ off,
                           const float* __restrict__ mlog,
                           const float* __restrict__ val,
                           _Float16* __restrict__ dcnH, int K) {
    int i = blockIdx.x * blockDim.x + threadIdx.x;
    if (i >= TOK * DIM) return;
    int t = i >> 6, c = i & 63;
    int b = t >> 10, l = t & 1023;
    const float* o = off  + (size_t)t * DIM * K + c * K;
    const float* m = mlog + (size_t)t * DIM * K + c * K;
    float mx = -3.0e38f;
    for (int k = 0; k < K; ++k) mx = fmaxf(mx, m[k]);
    float ex[17], se = 0.f;
    for (int k = 0; k < K; ++k) { ex[k] = expf(m[k] - mx); se += ex[k]; }
    float inv = 1.0f / se;
    int center = (K - 1) >> 1;
    const float* vb = val + (size_t)b * LSEQ * DIM + c;   // val[(b*L+l')*64 + c]
    float acc = 0.f;
    for (int k = 0; k < K; ++k) {
        float pos = (float)(l + k - center) + o[k];
        float p0 = floorf(pos);
        float w1 = pos - p0, w0 = 1.0f - w1;
        int i0 = (int)p0, i1 = i0 + 1;
        float v0 = (i0 >= 0 && i0 < LSEQ) ? vb[(size_t)i0 * DIM] : 0.0f;
        float v1 = (i1 >= 0 && i1 < LSEQ) ? vb[(size_t)i1 * DIM] : 0.0f;
        acc += (ex[k] * inv) * (w0 * v0 + w1 * v1);
    }
    dcnH[i] = (_Float16)acc;
}

__global__ void mul_f16(const float* __restrict__ a, const float* __restrict__ v,
                        _Float16* __restrict__ m, int n) {
    int i = blockIdx.x * blockDim.x + threadIdx.x;
    if (i < n) m[i] = (_Float16)(a[i] * v[i]);
}

// layer-scale residual + LayerNorm(channel) -> x1 (f32) and LN output (f16).
// one wave32 per token, 2 channels/lane, shuffle reductions.
__global__ __launch_bounds__(256)
void ls_ln(const float* __restrict__ xt, const float* __restrict__ res,
           const float* __restrict__ ls, const float* __restrict__ lw,
           const float* __restrict__ lb, float* __restrict__ x1,
           _Float16* __restrict__ hH) {
    int lane = threadIdx.x & 31;
    int t = blockIdx.x * 8 + (threadIdx.x >> 5);
    int i0 = t * DIM + lane, i1 = i0 + 32;
    float v0 = xt[i0] + ls[lane]      * res[i0];
    float v1 = xt[i1] + ls[lane + 32] * res[i1];
    x1[i0] = v0; x1[i1] = v1;
    float s = v0 + v1;
#pragma unroll
    for (int d = 16; d; d >>= 1) s += __shfl_xor(s, d, 32);
    float mu = s * (1.0f / 64.0f);
    float d0 = v0 - mu, d1 = v1 - mu;
    float q = d0 * d0 + d1 * d1;
#pragma unroll
    for (int d = 16; d; d >>= 1) q += __shfl_xor(q, d, 32);
    float rstd = rsqrtf(q * (1.0f / 64.0f) + 1e-6f);
    hH[i0] = (_Float16)(d0 * rstd * lw[lane]      + lb[lane]);
    hH[i1] = (_Float16)(d1 * rstd * lw[lane + 32] + lb[lane + 32]);
}

// ------------------------------- launcher ----------------------------------
extern "C" void kernel_launch(void* const* d_in, const int* in_sizes, int n_in,
                              void* d_out, int out_size, void* d_ws, size_t ws_size,
                              hipStream_t stream) {
    (void)in_sizes; (void)n_in; (void)out_size; (void)ws_size;
    // input leaf order (pytree flatten of setup_inputs):
    // 0:x, 1..6:a_pw_w, 7..12:a_pw_b, 13..18:off_w, 19..24:off_b,
    // 25..30:mask_w, 31..36:mask_b, 37..42:in_w, 43..48:in_b,
    // 49..54:out_w, 55..60:out_b, 61..66:v_w, 67..72:v_b,
    // 73..78:proj_w, 79..84:proj_b, 85:fc1_w, 86:fc1_b, 87:fc2_w, 88:fc2_b,
    // 89:ln_w, 90:ln_b, 91:layer_scale, 92:gamma2
    const float* X     = (const float*)d_in[0];
    const float* FC1_W = (const float*)d_in[85];
    const float* FC1_B = (const float*)d_in[86];
    const float* FC2_W = (const float*)d_in[87];
    const float* FC2_B = (const float*)d_in[88];
    const float* LN_W  = (const float*)d_in[89];
    const float* LN_B  = (const float*)d_in[90];
    const float* LS    = (const float*)d_in[91];
    const float* G2    = (const float*)d_in[92];
    float* OUT = (float*)d_out;

    // ---- workspace carve (peak ~93 MB) ----
    char* ws = (char*)d_ws;
    size_t cur = 0;
    auto alloc = [&](size_t bytes) -> void* {
        void* p = ws + cur;
        cur = (cur + bytes + 255) & ~(size_t)255;
        return p;
    };
    float*    xt     = (float*)   alloc((size_t)TOK * DIM * 4);
    _Float16* xth    = (_Float16*)alloc((size_t)TOK * DIM * 2);
    float*    x1     = (float*)   alloc((size_t)TOK * DIM * 4);
    float*    res    = (float*)   alloc((size_t)TOK * DIM * 4);
    _Float16* a_h    = (_Float16*)alloc((size_t)TOK * DIM * 2);
    float*    valb   = (float*)   alloc((size_t)TOK * DIM * 4);
    _Float16* dcn_h  = (_Float16*)alloc((size_t)TOK * DIM * 2);
    float*    aoutb  = (float*)   alloc((size_t)TOK * DIM * 4);
    float*    vbuf   = (float*)   alloc((size_t)TOK * DIM * 4);
    _Float16* m_h    = (_Float16*)alloc((size_t)TOK * DIM * 2);
    _Float16* h_h    = (_Float16*)alloc((size_t)TOK * DIM * 2);
    _Float16* h1_h   = (_Float16*)alloc((size_t)TOK * MLPH * 2);
    float*    offbuf = (float*)   alloc((size_t)TOK * DIM * 17 * 4);
    float*    mlogb  = (float*)   alloc((size_t)TOK * DIM * 17 * 4);
    _Float16 *aw_h[NPB], *vw_h[NPB], *pw_h[NPB], *inw_h[NPB], *outw_h[NPB];
    _Float16 *offw_h[NPB], *maskw_h[NPB];
    for (int i = 0; i < NPB; ++i) {
        int K = 7 + 2 * i;
        aw_h[i]    = (_Float16*)alloc(DIM * DIM * 2);
        vw_h[i]    = (_Float16*)alloc(DIM * DIM * 2);
        pw_h[i]    = (_Float16*)alloc(DIM * DIM * 2);
        inw_h[i]   = (_Float16*)alloc(DIM * DIM * 2);
        outw_h[i]  = (_Float16*)alloc(DIM * DIM * 2);
        offw_h[i]  = (_Float16*)alloc((size_t)DIM * DIM * K * 2);
        maskw_h[i] = (_Float16*)alloc((size_t)DIM * DIM * K * 2);
    }
    _Float16* fc1_h = (_Float16*)alloc(DIM * MLPH * 2);
    _Float16* fc2_h = (_Float16*)alloc(DIM * MLPH * 2);

    auto blocks = [](int n) { return (n + 255) / 256; };

    // ---- stage 0: residual accumulator, x transpose, weight prep ----
    hipMemsetAsync(res, 0, (size_t)TOK * DIM * 4, stream);
    prep_x<<<blocks(TOK * DIM), 256, 0, stream>>>(X, xt, xth);
    for (int i = 0; i < NPB; ++i) {
        int K = 7 + 2 * i, NK = DIM * K;
        cvt_f16<<<blocks(DIM * DIM), 256, 0, stream>>>((const float*)d_in[1 + i],  aw_h[i], DIM * DIM);
        cvt_f16<<<blocks(DIM * DIM), 256, 0, stream>>>((const float*)d_in[61 + i], vw_h[i], DIM * DIM);
        cvt_f16<<<blocks(DIM * DIM), 256, 0, stream>>>((const float*)d_in[73 + i], pw_h[i], DIM * DIM);
        tr_f16<<<blocks(DIM * DIM), 256, 0, stream>>>((const float*)d_in[37 + i], inw_h[i],  DIM, DIM);
        tr_f16<<<blocks(DIM * DIM), 256, 0, stream>>>((const float*)d_in[49 + i], outw_h[i], DIM, DIM);
        tr_f16<<<blocks(DIM * NK), 256, 0, stream>>>((const float*)d_in[13 + i], offw_h[i],  DIM, NK);
        tr_f16<<<blocks(DIM * NK), 256, 0, stream>>>((const float*)d_in[25 + i], maskw_h[i], DIM, NK);
    }
    tr_f16<<<blocks(DIM * MLPH), 256, 0, stream>>>(FC1_W, fc1_h, DIM,  MLPH);
    tr_f16<<<blocks(DIM * MLPH), 256, 0, stream>>>(FC2_W, fc2_h, MLPH, DIM);

    // ---- stage 1: six deformable branches ----
    for (int i = 0; i < NPB; ++i) {
        int K = 7 + 2 * i, NK = DIM * K;
        const float* apw_b  = (const float*)d_in[7 + i];
        const float* off_b  = (const float*)d_in[19 + i];
        const float* mask_b = (const float*)d_in[31 + i];
        const float* in_b   = (const float*)d_in[43 + i];
        const float* out_b  = (const float*)d_in[55 + i];
        const float* v_b    = (const float*)d_in[67 + i];
        const float* proj_b = (const float*)d_in[79 + i];

        // a = GELU(a_pw_w @ x + b)  -> f16
        gemm_wmma<EPI_GELU_F16><<<dim3(TOK / 64, 1), 128, 0, stream>>>(
            xth, aw_h[i], apw_b, DIM, DIM, nullptr, a_h, nullptr, nullptr);
        // offset / mask-logit / value GEMMs (the FLOP-dominant part)
        gemm_wmma<EPI_F32><<<dim3(TOK / 64, NK / 64), 128, 0, stream>>>(
            a_h, offw_h[i], off_b, NK, DIM, offbuf, nullptr, nullptr, nullptr);
        gemm_wmma<EPI_F32><<<dim3(TOK / 64, NK / 64), 128, 0, stream>>>(
            a_h, maskw_h[i], mask_b, NK, DIM, mlogb, nullptr, nullptr, nullptr);
        gemm_wmma<EPI_F32><<<dim3(TOK / 64, 1), 128, 0, stream>>>(
            a_h, inw_h[i], in_b, DIM, DIM, valb, nullptr, nullptr, nullptr);
        // softmax + bilinear deformable sampling
        dcn_sample<<<blocks(TOK * DIM), 256, 0, stream>>>(offbuf, mlogb, valb, dcn_h, K);
        // out-proj, v branch, a*v, proj accumulate into res
        gemm_wmma<EPI_F32><<<dim3(TOK / 64, 1), 128, 0, stream>>>(
            dcn_h, outw_h[i], out_b, DIM, DIM, aoutb, nullptr, nullptr, nullptr);
        gemm_wmma<EPI_F32><<<dim3(TOK / 64, 1), 128, 0, stream>>>(
            xth, vw_h[i], v_b, DIM, DIM, vbuf, nullptr, nullptr, nullptr);
        mul_f16<<<blocks(TOK * DIM), 256, 0, stream>>>(aoutb, vbuf, m_h, TOK * DIM);
        gemm_wmma<EPI_ACC><<<dim3(TOK / 64, 1), 128, 0, stream>>>(
            m_h, pw_h[i], proj_b, DIM, DIM, res, nullptr, nullptr, nullptr);
    }

    // ---- stage 2: layer-scale residual + LN + MLP + final residual ----
    ls_ln<<<TOK / 8, 256, 0, stream>>>(xt, res, LS, LN_W, LN_B, x1, h_h);
    gemm_wmma<EPI_GELU_F16><<<dim3(TOK / 64, MLPH / 64), 128, 0, stream>>>(
        h_h, fc1_h, FC1_B, MLPH, DIM, nullptr, h1_h, nullptr, nullptr);
    gemm_wmma<EPI_FINAL><<<dim3(TOK / 64, 1), 128, 0, stream>>>(
        h1_h, fc2_h, FC2_B, DIM, MLPH, OUT, nullptr, x1, G2);
}
